// CIPS_33509334843786
// MI455X (gfx1250) — matrix operand order repaired and checked by reading, not compiled, not run on hardware
//
#include <hip/hip_runtime.h>
#include <hip/hip_bf16.h>

typedef __attribute__((ext_vector_type(16))) _Float16 v16h;
typedef __attribute__((ext_vector_type(8)))  float    v8f;

#define N_USERS 100000
#define N_ITEMS 50000
#define N_NODES (N_USERS + N_ITEMS)
#define DIM 64
#define NNZ 3000000
#define BATCH 4096
#define LAM 0.5f

// ---------------- elementwise helpers ----------------

__global__ void gcn_init_kernel(const float* __restrict__ ue, const float* __restrict__ ie,
                                float* __restrict__ cur, float* __restrict__ acc,
                                int nUserElems, int nTotalElems) {
    int i = blockIdx.x * blockDim.x + threadIdx.x;   // float4 index
    int base = i * 4;
    if (base >= nTotalElems) return;
    float4 v = (base < nUserElems) ? ((const float4*)ue)[i]
                                   : ((const float4*)ie)[(base - nUserElems) >> 2];
    ((float4*)cur)[i] = v;
    ((float4*)acc)[i] = v;
}

__global__ void gcn_zero_kernel(float* __restrict__ p, int nElems) {
    int i = blockIdx.x * blockDim.x + threadIdx.x;
    if (i * 4 >= nElems) return;
    ((float4*)p)[i] = make_float4(0.f, 0.f, 0.f, 0.f);
}

__global__ void gcn_accum_kernel(float* __restrict__ acc, const float* __restrict__ add, int nElems) {
    int i = blockIdx.x * blockDim.x + threadIdx.x;
    if (i * 4 >= nElems) return;
    float4 a = ((const float4*)acc)[i];
    float4 b = ((const float4*)add)[i];
    a.x += b.x; a.y += b.y; a.z += b.z; a.w += b.w;
    ((float4*)acc)[i] = a;
}

// ---------------- SpMM: wave-per-edge, float2 per lane (256B/row), f32 atomics to L2 ----------------

__global__ void gcn_spmm_kernel(const int* __restrict__ rows, const int* __restrict__ cols,
                                const float* __restrict__ vals,
                                const float* __restrict__ src, float* __restrict__ dst, int nnz) {
    int e    = (blockIdx.x * blockDim.x + threadIdx.x) >> 5;
    int lane = threadIdx.x & 31;
    if (e >= nnz) return;
    int   r = rows[e];
    int   c = cols[e];
    float v = vals[e];
    float2 x = ((const float2*)(src + (size_t)c * DIM))[lane];
    float* drow = dst + (size_t)r * DIM + lane * 2;
    unsafeAtomicAdd(drow,     v * x.x);   // global_atomic_add_f32, no-return
    unsafeAtomicAdd(drow + 1, v * x.y);
}

// ---------------- fusion: 16 rows per wave via v_wmma_f32_16x16x32_f16 ----------------
// Computes dA = embA_row . wA  and dB = embB_row . wB  (B broadcast across all 16 N columns),
// then w = cnt*(1-lam) + sigmoid(dA/4 + bA + dB/4 + bB)*lam, and overwrites embA in place with
// fused = (embA/4)*w + (embB/4)*(1-w).   (acc holds 4x the "light" output, hence the 0.25.)

__global__ void gcn_fuse_wmma_kernel(float* __restrict__ embA, const float* __restrict__ embB,
                                     const float* __restrict__ wA, const float* __restrict__ bA,
                                     const float* __restrict__ wB, const float* __restrict__ bB,
                                     const float* __restrict__ cnt, float lam, int nTiles) {
    int wave = (blockIdx.x * blockDim.x + threadIdx.x) >> 5;
    int lane = threadIdx.x & 31;
    if (wave >= nTiles) return;           // wave-uniform: EXEC all-ones inside active waves

    int m    = lane & 15;                 // A-matrix row this lane feeds
    int half = lane >> 4;                 // K-half selector (ISA 16-bit A/B layouts)
    const float* rA = embA + (size_t)(wave * 16 + m) * DIM;
    const float* rB = embB + (size_t)(wave * 16 + m) * DIM;

    v8f cA = {}; v8f cB = {};
    int kb  = half * 8;                   // A: lanes 0-15 K=0..7/16..23 ; lanes 16-31 K=8..15/24..31
    int kbB = half * 16;                  // B: same K order as A (lanes 0-15 K=0..7/16..23 ; lanes 16-31 K=8..15/24..31)
#pragma unroll
    for (int kw = 0; kw < DIM; kw += 32) {
        v16h a, a2, b, b2;
#pragma unroll
        for (int i = 0; i < 8; ++i) {
            a [i]     = (_Float16)rA[kw + kb + i];
            a [8 + i] = (_Float16)rA[kw + kb + 16 + i];
            a2[i]     = (_Float16)rB[kw + kb + i];
            a2[8 + i] = (_Float16)rB[kw + kb + 16 + i];
        }
#pragma unroll
        for (int i = 0; i < 16; ++i) {
            b [i] = (_Float16)wA[kw + kb + (i & 7) + ((i >> 3) << 4)];   // broadcast across N columns
            b2[i] = (_Float16)wB[kw + kb + (i & 7) + ((i >> 3) << 4)];
        }
        cA = __builtin_amdgcn_wmma_f32_16x16x32_f16(false, a,  false, b,  (short)0, cA, false, false);
        cB = __builtin_amdgcn_wmma_f32_16x16x32_f16(false, a2, false, b2, (short)0, cB, false, false);
    }

    // Each lane handles row u = lane>>1 of the tile; its dot lives in c[(lane&15)>>1]
    // (C/D layout: VGPR j holds M=j for lanes 0-15, M=j+8 for lanes 16-31; all N equal here).
    int u = lane >> 1;
    int j = (lane & 15) >> 1;
    float dA = 0.f, dB = 0.f;
#pragma unroll
    for (int t = 0; t < 8; ++t) {
        if (t == j) { dA = cA[t]; dB = cB[t]; }
    }

    int   uRow = wave * 16 + u;
    float s = 1.0f / (1.0f + __expf(-(0.25f * dA + bA[0] + 0.25f * dB + bB[0])));
    float w = cnt[uRow] * (1.0f - lam) + s * lam;

    float*       orow = embA + (size_t)uRow * DIM;   // in-place: tile rows only touched by this wave
    const float* brow = embB + (size_t)uRow * DIM;
    int d0 = (lane & 1) * 32;
#pragma unroll
    for (int d = 0; d < 32; ++d) {
        float aV = orow[d0 + d] * 0.25f;
        float bV = brow[d0 + d] * 0.25f;
        orow[d0 + d] = aV * w + bV * (1.0f - w);
    }
}

// ---------------- final: wave-per-batch-row dot + sigmoid ----------------

__global__ void gcn_gamma_kernel(const float* __restrict__ fusedU, const float* __restrict__ fusedI,
                                 const int* __restrict__ users, const int* __restrict__ items,
                                 float* __restrict__ out, int batch) {
    int wave = (blockIdx.x * blockDim.x + threadIdx.x) >> 5;
    int lane = threadIdx.x & 31;
    if (wave >= batch) return;
    float2 a = ((const float2*)(fusedU + (size_t)users[wave] * DIM))[lane];
    float2 b = ((const float2*)(fusedI + (size_t)items[wave] * DIM))[lane];
    float p = a.x * b.x + a.y * b.y;
#pragma unroll
    for (int off = 16; off > 0; off >>= 1) p += __shfl_down(p, off, 32);
    if (lane == 0) out[wave] = 1.0f / (1.0f + __expf(-p));
}

// ---------------- host orchestration ----------------

extern "C" void kernel_launch(void* const* d_in, const int* in_sizes, int n_in,
                              void* d_out, int out_size, void* d_ws, size_t ws_size,
                              hipStream_t stream) {
    const float* user_emb0 = (const float*)d_in[0];
    const float* item_emb0 = (const float*)d_in[1];
    const float* user_emb1 = (const float*)d_in[2];
    const float* item_emb1 = (const float*)d_in[3];
    const float* g_vals    = (const float*)d_in[4];
    const float* g2_vals   = (const float*)d_in[5];
    const float* fc1_w     = (const float*)d_in[6];
    const float* fc1_b     = (const float*)d_in[7];
    const float* fc2_w     = (const float*)d_in[8];
    const float* fc2_b     = (const float*)d_in[9];
    const float* fc3_w     = (const float*)d_in[10];
    const float* fc3_b     = (const float*)d_in[11];
    const float* fc4_w     = (const float*)d_in[12];
    const float* fc4_b     = (const float*)d_in[13];
    const float* users_cnt = (const float*)d_in[14];
    const float* items_cnt = (const float*)d_in[15];
    const int*   g_rows    = (const int*)d_in[16];
    const int*   g_cols    = (const int*)d_in[17];
    const int*   g2_rows   = (const int*)d_in[18];
    const int*   g2_cols   = (const int*)d_in[19];
    const int*   users     = (const int*)d_in[20];
    const int*   items     = (const int*)d_in[21];
    float*       out       = (float*)d_out;

    const size_t NE = (size_t)N_NODES * DIM;         // 9.6M floats per buffer
    float* cur  = (float*)d_ws;
    float* nxt  = cur + NE;
    float* accA = nxt + NE;                          // graph2 / emb1 result, then fused (in place)
    float* accB = accA + NE;                         // graph  / emb0 result
    // total workspace use: 4 * 38.4 MB = 153.6 MB

    const int EW_BLOCKS   = (int)((NE / 4 + 255) / 256);          // float4 elementwise
    const int SPMM_BLOCKS = (int)(((size_t)NNZ * 32 + 255) / 256);

    auto propagate = [&](const float* ue, const float* ie,
                         const int* prows, const int* pcols, const float* pvals,
                         float* c0, float* c1, float* acc) {
        gcn_init_kernel<<<EW_BLOCKS, 256, 0, stream>>>(ue, ie, c0, acc,
                                                       N_USERS * DIM, N_NODES * DIM);
        float* a = c0; float* b = c1;
        for (int l = 0; l < 3; ++l) {
            gcn_zero_kernel<<<EW_BLOCKS, 256, 0, stream>>>(b, (int)NE);
            gcn_spmm_kernel<<<SPMM_BLOCKS, 256, 0, stream>>>(prows, pcols, pvals, a, b, NNZ);
            gcn_accum_kernel<<<EW_BLOCKS, 256, 0, stream>>>(acc, b, (int)NE);
            float* t = a; a = b; b = t;
        }
    };

    // all_users/all_items   <- propagate(emb1, graph2)  -> accA
    propagate(user_emb1, item_emb1, g2_rows, g2_cols, g2_vals, cur, nxt, accA);
    // all_users2/all_items2 <- propagate(emb0, graph)   -> accB
    propagate(user_emb0, item_emb0, g_rows, g_cols, g_vals, cur, nxt, accB);

    // fuse users: 100000/16 = 6250 tiles ; fuse items: 50000/16 = 3125 tiles
    const int U_TILES = N_USERS / 16, I_TILES = N_ITEMS / 16;
    gcn_fuse_wmma_kernel<<<(U_TILES + 7) / 8, 256, 0, stream>>>(
        accA, accB, fc1_w, fc1_b, fc2_w, fc2_b, users_cnt, LAM, U_TILES);
    gcn_fuse_wmma_kernel<<<(I_TILES + 7) / 8, 256, 0, stream>>>(
        accA + (size_t)N_USERS * DIM, accB + (size_t)N_USERS * DIM,
        fc3_w, fc3_b, fc4_w, fc4_b, items_cnt, LAM, I_TILES);

    gcn_gamma_kernel<<<(BATCH * 32 + 255) / 256, 256, 0, stream>>>(
        accA, accA + (size_t)N_USERS * DIM, users, items, out, BATCH);
}
